// CentroidPool_42322607734953
// MI455X (gfx1250) — compile-verified, matrix operand order checked by reading, not run on hardware
//
#include <hip/hip_runtime.h>
#include <hip/hip_bf16.h>

typedef __attribute__((ext_vector_type(8)))  float    v8f;
typedef __attribute__((ext_vector_type(8)))  _Float16 v8h;
typedef __attribute__((ext_vector_type(16))) _Float16 v16h;

#define D_DIM   128
#define TILE_C  16
#define ROW_PAD 136   // padded LDS row stride (272B): lanes stripe over all 64 banks

// --- prep kernel: split coords into f16 hi/lo and compute squared norms ---
__global__ void coords_prep_kernel(const float* __restrict__ coords,
                                   _Float16* __restrict__ cHi,
                                   _Float16* __restrict__ cLo,
                                   float* __restrict__ c2, int K) {
    int c = blockIdx.x * blockDim.x + threadIdx.x;
    if (c >= K) return;
    const float* row = coords + (size_t)c * D_DIM;
    _Float16* hRow = cHi + (size_t)c * D_DIM;
    _Float16* lRow = cLo + (size_t)c * D_DIM;
    float s = 0.f;
#pragma unroll 8
    for (int i = 0; i < D_DIM; ++i) {
        float x = row[i];
        s += x * x;
        _Float16 hi = (_Float16)x;             // RNE, 11-bit mantissa
        hRow[i] = hi;
        lRow[i] = (_Float16)(x - (float)hi);   // residual -> ~22 combined bits
    }
    c2[c] = s;
}

// --- main kernel: f16x3 split-precision WMMA GEMM + fused argmin ---
// One wave per 16-row tile; 8 waves (256 threads) per block -> 128 rows/block.
// B tiles staged in LDS via async global->LDS copies, double buffered.
__global__ __launch_bounds__(256)
void centroid_argmin_kernel(const float* __restrict__ latent,
                            const _Float16* __restrict__ coordsHi,
                            const _Float16* __restrict__ coordsLo,
                            const float* __restrict__ c2,
                            int* __restrict__ out, int K) {
    __shared__ _Float16 tile[2][2][TILE_C * ROW_PAD];  // [buf][hi/lo][16 rows, padded]

    const int tid  = threadIdx.x;
    const int lane = tid & 31;
    const int wave = tid >> 5;
    const int r    = lane & 15;   // row (A) / col (B) within tile
    const int h    = lane >> 4;   // lane half: K groups 8h+0..7 and 16+8h+0..7
    const int rowBase = blockIdx.x * 128 + wave * 16;

    // Cooperative staging role: 256 threads x (16B hi + 16B lo) = 8KB tile.
    const int sRow   = tid >> 4;  // 0..15 centroid row within tile
    const int sChunk = tid & 15;  // 0..15 16-byte chunk within row

    // Load the wave's 16x128 fp32 A-tile once; split into f16 hi/lo fragments.
    // 16-bit A 16x32 layout: v16 elems 0..7 = K 8h+0..7, elems 8..15 = K 16+8h+0..7.
    const float* aRow = latent + (size_t)(rowBase + r) * D_DIM;
    v16h aHi[4], aLo[4];
#pragma unroll
    for (int kc = 0; kc < 4; ++kc) {
        const int b0 = kc * 32 + 8 * h;
        const int b1 = b0 + 16;
        float xs[16];
        *(float4*)(&xs[0])  = *(const float4*)(aRow + b0);
        *(float4*)(&xs[4])  = *(const float4*)(aRow + b0 + 4);
        *(float4*)(&xs[8])  = *(const float4*)(aRow + b1);
        *(float4*)(&xs[12]) = *(const float4*)(aRow + b1 + 4);
#pragma unroll
        for (int i = 0; i < 16; ++i) {
            _Float16 hi = (_Float16)xs[i];
            aHi[kc][i] = hi;
            aLo[kc][i] = (_Float16)(xs[i] - (float)hi);
        }
    }

    float best[8];
    int   bidx[8];
#pragma unroll
    for (int v = 0; v < 8; ++v) { best[v] = 3.4e38f; bidx[v] = 0; }

    const int numTiles = K / TILE_C;

    // Async-copy issue helper: this thread's hi+lo 16B chunks of tile `ct` -> buf.
    auto issue_tile = [&](int ct, int buf) {
        const size_t goff = (size_t)(ct * TILE_C + sRow) * D_DIM + sChunk * 8;
        unsigned long long gHi = (unsigned long long)(uintptr_t)(coordsHi + goff);
        unsigned long long gLo = (unsigned long long)(uintptr_t)(coordsLo + goff);
        unsigned dHi = (unsigned)(uintptr_t)&tile[buf][0][sRow * ROW_PAD + sChunk * 8];
        unsigned dLo = (unsigned)(uintptr_t)&tile[buf][1][sRow * ROW_PAD + sChunk * 8];
        asm volatile("global_load_async_to_lds_b128 %0, %1, off"
                     :: "v"(dHi), "v"(gHi) : "memory");
        asm volatile("global_load_async_to_lds_b128 %0, %1, off"
                     :: "v"(dLo), "v"(gLo) : "memory");
    };

    issue_tile(0, 0);  // prologue: tile 0 in flight

#pragma unroll 1
    for (int ct = 0; ct < numTiles; ++ct) {
        const int buf = ct & 1;
        if (ct + 1 < numTiles) {
            issue_tile(ct + 1, buf ^ 1);                       // prefetch next tile
            asm volatile("s_wait_asynccnt 0x2" ::: "memory");  // tile ct landed
        } else {
            asm volatile("s_wait_asynccnt 0x0" ::: "memory");
        }
        __syncthreads();  // publish tile ct to all waves

        const int col = ct * TILE_C + r;
        const float cc = c2[col];  // issued early; wait covered by wmma burst

        // Load ALL B fragments first (distinct registers) -> one ds clause +
        // one s_wait_dscnt, then an uninterrupted 12-wmma burst.
        const _Float16* bhRow = &tile[buf][0][r * ROW_PAD];
        const _Float16* blRow = &tile[buf][1][r * ROW_PAD];
        v16h Bh[4], Bl[4];
#pragma unroll
        for (int kc = 0; kc < 4; ++kc) {
            const int b0 = kc * 32 + 8 * h;
            const int b1 = b0 + 16;
            v8h bh0 = *(const v8h*)(bhRow + b0);
            v8h bh1 = *(const v8h*)(bhRow + b1);
            v8h bl0 = *(const v8h*)(blRow + b0);
            v8h bl1 = *(const v8h*)(blRow + b1);
            Bh[kc] = __builtin_shufflevector(bh0, bh1,
                0,1,2,3,4,5,6,7,8,9,10,11,12,13,14,15);
            Bl[kc] = __builtin_shufflevector(bl0, bl1,
                0,1,2,3,4,5,6,7,8,9,10,11,12,13,14,15);
        }

        // 4 independent 3-deep accumulator chains -> XDL stays issue-limited.
        v8f accs[4];
#pragma unroll
        for (int kc = 0; kc < 4; ++kc)
            accs[kc] = (v8f){0.f, 0.f, 0.f, 0.f, 0.f, 0.f, 0.f, 0.f};
#pragma unroll
        for (int kc = 0; kc < 4; ++kc) {
            accs[kc] = __builtin_amdgcn_wmma_f32_16x16x32_f16(
                false, aHi[kc], false, Bh[kc], (short)0, accs[kc], false, false);
            accs[kc] = __builtin_amdgcn_wmma_f32_16x16x32_f16(
                false, aLo[kc], false, Bh[kc], (short)0, accs[kc], false, false);
            accs[kc] = __builtin_amdgcn_wmma_f32_16x16x32_f16(
                false, aHi[kc], false, Bl[kc], (short)0, accs[kc], false, false);
        }

#pragma unroll
        for (int v = 0; v < 8; ++v) {
            float dot = (accs[0][v] + accs[1][v]) + (accs[2][v] + accs[3][v]);
            float score = fmaf(-2.f, dot, cc);    // ||c||^2 - 2 x.c
            if (score < best[v]) { best[v] = score; bidx[v] = col; }
        }
        __syncthreads();  // all waves done reading buf before it is refilled
    }

    // Argmin across the 16 lanes of each half (columns live across lanes).
    // Tie-break: lower centroid index, matching argmin's first-minimum rule.
#pragma unroll
    for (int m = 1; m <= 8; m <<= 1) {
#pragma unroll
        for (int v = 0; v < 8; ++v) {
            float os = __shfl_xor(best[v], m, 32);
            int   oi = __shfl_xor(bidx[v], m, 32);
            if (os < best[v] || (os == best[v] && oi < bidx[v])) {
                best[v] = os; bidx[v] = oi;
            }
        }
    }

    // D layout: slot v => row v (lanes 0-15) or row v+8 (lanes 16-31).
#pragma unroll
    for (int v = 0; v < 8; ++v) {
        if (r == v) out[rowBase + v + 8 * h] = bidx[v];
    }
}

extern "C" void kernel_launch(void* const* d_in, const int* in_sizes, int n_in,
                              void* d_out, int out_size, void* d_ws, size_t ws_size,
                              hipStream_t stream) {
    const float* latent = (const float*)d_in[0];
    const float* coords = (const float*)d_in[1];
    const int N = in_sizes[0] / D_DIM;   // 131072
    const int K = in_sizes[1] / D_DIM;   // 1024

    // d_ws layout: [c2: K fp32][coordsHi: K*128 f16][coordsLo: K*128 f16]
    char* ws = (char*)d_ws;
    float*    c2  = (float*)ws;
    _Float16* cHi = (_Float16*)(ws + (size_t)K * sizeof(float));
    _Float16* cLo = (_Float16*)(ws + (size_t)K * sizeof(float)
                                   + (size_t)K * D_DIM * sizeof(_Float16));
    int* out = (int*)d_out;              // argmin indices (int32 output dtype)

    coords_prep_kernel<<<(K + 255) / 256, 256, 0, stream>>>(coords, cHi, cLo, c2, K);
    centroid_argmin_kernel<<<N / 128, 256, 0, stream>>>(latent, cHi, cLo, c2, out, K);
}
